// PairFormerIteration_33913061769437
// MI455X (gfx1250) — matrix operand order repaired and checked by reading, not compiled, not run on hardware
//
#include <hip/hip_runtime.h>

typedef __bf16 bf16_t;
typedef __attribute__((ext_vector_type(16))) __bf16 v16bf;
typedef __attribute__((ext_vector_type(8))) float v8f;

#define RN 65536      // N*N rows
#define NSEQ 256
#define CDIM 128

#define USE_ASYNC 1

// ---------------------------------------------------------------- utilities
__device__ __forceinline__ float sigmoidf_(float x) { return 1.0f / (1.0f + __expf(-x)); }

__device__ __forceinline__ v8f wmma_bf16(v16bf a, v16bf b, v8f c) {
  // D = A(16x32) * B(32x16) + C, f32 accum
  return __builtin_amdgcn_wmma_f32_16x16x32_bf16(false, a, false, b, (short)0, c, false, false);
}

// async 16B copy global -> LDS (ASYNCcnt-tracked, no VGPR round trip)
__device__ __forceinline__ void async_copy_b128(void* lds_dst, const void* gsrc) {
#if USE_ASYNC
  unsigned ldsoff = (unsigned)(unsigned long long)lds_dst;  // low 32 bits = LDS offset
  unsigned long long ga = (unsigned long long)gsrc;
  asm volatile("global_load_async_to_lds_b128 %0, %1, off" ::"v"(ldsoff), "v"(ga) : "memory");
#else
  *(uint4*)lds_dst = *(const uint4*)gsrc;
#endif
}
__device__ __forceinline__ void async_wait() {
#if USE_ASYNC
  asm volatile("s_wait_asynccnt 0x0" ::: "memory");
#endif
}

// A fragment (16x32, MxK): lanes 0-15 -> M=lane, K={0..7,16..23}; lanes 16-31 -> K={8..15,24..31}
__device__ __forceinline__ v16bf frag_a(const bf16_t* base, int ld, int m0, int k0) {
  int ln = threadIdx.x & 31;
  const bf16_t* row = base + (size_t)(m0 + (ln & 15)) * ld + k0 + ((ln >> 4) * 8);
  v16bf a;
#pragma unroll
  for (int t = 0; t < 8; ++t) a[t] = row[t];
#pragma unroll
  for (int t = 0; t < 8; ++t) a[8 + t] = row[16 + t];
  return a;
}

// B fragment (32x16, KxN): lane n = ln&15; lanes 0-15 -> K={0..7,16..23}; lanes 16-31 -> K={8..15,24..31}
__device__ __forceinline__ v16bf frag_b(const bf16_t* base, int ld, int n0, int k0) {
  int ln = threadIdx.x & 31;
  const bf16_t* row = base + (size_t)(n0 + (ln & 15)) * ld + k0 + ((ln >> 4) * 8);
  v16bf b;
#pragma unroll
  for (int t = 0; t < 8; ++t) b[t] = row[t];
#pragma unroll
  for (int t = 0; t < 8; ++t) b[8 + t] = row[16 + t];
  return b;
}

// ---------------------------------------------------------------- LayerNorm -> bf16 (wave per 128-wide row)
__global__ __launch_bounds__(256) void k_ln(const float* __restrict__ src,
                                            const float* __restrict__ sc,
                                            const float* __restrict__ bi,
                                            bf16_t* __restrict__ dst) {
  int row = blockIdx.x * 8 + (threadIdx.x >> 5);
  int ln = threadIdx.x & 31;
  const float4 v = *(const float4*)(src + (size_t)row * CDIM + ln * 4);
  float s = v.x + v.y + v.z + v.w;
#pragma unroll
  for (int off = 16; off > 0; off >>= 1) s += __shfl_xor(s, off, 32);
  float mu = s * (1.0f / 128.0f);
  float dx = v.x - mu, dy = v.y - mu, dz = v.z - mu, dw = v.w - mu;
  float s2 = dx * dx + dy * dy + dz * dz + dw * dw;
#pragma unroll
  for (int off = 16; off > 0; off >>= 1) s2 += __shfl_xor(s2, off, 32);
  float rs = rsqrtf(s2 * (1.0f / 128.0f) + 1e-5f);
  int c = ln * 4;
  bf16_t o[4];
  o[0] = (bf16_t)(dx * rs * sc[c + 0] + bi[c + 0]);
  o[1] = (bf16_t)(dy * rs * sc[c + 1] + bi[c + 1]);
  o[2] = (bf16_t)(dz * rs * sc[c + 2] + bi[c + 2]);
  o[3] = (bf16_t)(dw * rs * sc[c + 3] + bi[c + 3]);
  *(uint2*)(dst + (size_t)row * CDIM + c) = *(uint2*)o;
}

// ---------------------------------------------------------------- row transpose of 128-wide bf16 rows
__global__ __launch_bounds__(256) void k_transpose128(const bf16_t* __restrict__ src,
                                                      bf16_t* __restrict__ dst) {
  int r = blockIdx.x * 8 + (threadIdx.x >> 5);
  int ln = threadIdx.x & 31;
  int sr = ((r & 255) << 8) | (r >> 8);
  const uint2* s = (const uint2*)(src + (size_t)sr * CDIM);
  uint2* d = (uint2*)(dst + (size_t)r * CDIM);
  d[ln] = s[ln];
}

// ---------------------------------------------------------------- attention bias: bias[h][q][k] = x[q,k,:].Wb[h]
__global__ __launch_bounds__(256) void k_bias(const bf16_t* __restrict__ XB,
                                              const float* __restrict__ Wb,
                                              float* __restrict__ bias) {
  int gid = blockIdx.x * 256 + threadIdx.x;
  int h = gid & 3;
  int r = gid >> 2;
  const bf16_t* x = XB + (size_t)r * CDIM;
  const float* w = Wb + h * CDIM;
  float s = 0.f;
#pragma unroll 4
  for (int c = 0; c < CDIM; ++c) s += (float)x[c] * w[c];
  bias[(size_t)h * RN + r] = s;
}

// ---------------------------------------------------------------- O * gate -> bf16
__global__ __launch_bounds__(256) void k_og(const float* __restrict__ O,
                                            const float* __restrict__ G,
                                            bf16_t* __restrict__ OG) {
  size_t i = ((size_t)blockIdx.x * 256 + threadIdx.x) * 4;
  float4 o = *(const float4*)(O + i);
  float4 g = *(const float4*)(G + i);
  bf16_t t[4] = {(bf16_t)(o.x * g.x), (bf16_t)(o.y * g.y), (bf16_t)(o.z * g.z), (bf16_t)(o.w * g.w)};
  *(uint2*)(OG + i) = *(uint2*)t;
}

// ---------------------------------------------------------------- generic WMMA GEMM: out = X[M,K] * W^T (+epilogue)
// block tile 128x64, 8 waves in 4x2 grid, each wave owns 32x32 (2x2 WMMA tiles)
enum { EP_F32 = 0, EP_SIG = 1, EP_BF16 = 2, EP_ADD = 3, EP_GADD = 4 };

template <int WTRANS, int EP, int PERM>
__global__ __launch_bounds__(256) void k_gemm(const bf16_t* __restrict__ X,
                                              const float* __restrict__ W,
                                              float* __restrict__ dstF,
                                              bf16_t* __restrict__ dstB,
                                              const float* __restrict__ gate,
                                              int Nout, int K, float alpha) {
  __shared__ bf16_t As[128][48];
  __shared__ bf16_t Bs[64][48];
  int r0 = blockIdx.x * 128, n0 = blockIdx.y * 64;
  int tid = threadIdx.x, wave = tid >> 5, ln = tid & 31;
  int wy = (wave >> 1) * 32, wx = (wave & 1) * 32;
  v8f acc00 = {}, acc01 = {}, acc10 = {}, acc11 = {};
  for (int k0 = 0; k0 < K; k0 += 32) {
    __syncthreads();
    {  // A tile: 128 rows x 32 k, pure bf16 copy -> async to LDS
      int m = tid >> 1, seg = (tid & 1) * 16;
      const bf16_t* g = X + (size_t)(r0 + m) * K + k0 + seg;
      async_copy_b128(&As[m][seg], g);
      async_copy_b128(&As[m][seg + 8], g + 8);
    }
    if (WTRANS == 0) {  // W[Nout][K], convert f32->bf16
      int n = tid >> 2, seg = (tid & 3) * 8;
      const float* wsrc = W + (size_t)(n0 + n) * K + k0 + seg;
      float4 w0 = *(const float4*)wsrc;
      float4 w1 = *(const float4*)(wsrc + 4);
      bf16_t t8[8] = {(bf16_t)w0.x, (bf16_t)w0.y, (bf16_t)w0.z, (bf16_t)w0.w,
                      (bf16_t)w1.x, (bf16_t)w1.y, (bf16_t)w1.z, (bf16_t)w1.w};
      *(uint4*)&Bs[n][seg] = *(uint4*)t8;
    } else {  // W[K][Nout], transpose while converting
      int kk = tid >> 3, seg = (tid & 7) * 8;
      const float* wsrc = W + (size_t)(k0 + kk) * Nout + n0 + seg;
      float4 w0 = *(const float4*)wsrc;
      float4 w1 = *(const float4*)(wsrc + 4);
      Bs[seg + 0][kk] = (bf16_t)w0.x;
      Bs[seg + 1][kk] = (bf16_t)w0.y;
      Bs[seg + 2][kk] = (bf16_t)w0.z;
      Bs[seg + 3][kk] = (bf16_t)w0.w;
      Bs[seg + 4][kk] = (bf16_t)w1.x;
      Bs[seg + 5][kk] = (bf16_t)w1.y;
      Bs[seg + 6][kk] = (bf16_t)w1.z;
      Bs[seg + 7][kk] = (bf16_t)w1.w;
    }
    async_wait();
    __syncthreads();
    v16bf a0 = frag_a(&As[0][0], 48, wy, 0);
    v16bf a1 = frag_a(&As[0][0], 48, wy + 16, 0);
    v16bf b0 = frag_b(&Bs[0][0], 48, wx, 0);
    v16bf b1 = frag_b(&Bs[0][0], 48, wx + 16, 0);
    acc00 = wmma_bf16(a0, b0, acc00);
    acc01 = wmma_bf16(a0, b1, acc01);
    acc10 = wmma_bf16(a1, b0, acc10);
    acc11 = wmma_bf16(a1, b1, acc11);
  }
  int nloc = ln & 15, mh = (ln >> 4) * 8;
  auto emit = [&](const v8f& acc, int mt, int nt) {
    int col = n0 + wx + nt + nloc;
#pragma unroll
    for (int r = 0; r < 8; ++r) {
      int row = r0 + wy + mt + mh + r;
      float val = acc[r] * alpha;
      if (EP == EP_F32) dstF[(size_t)row * Nout + col] = val;
      if (EP == EP_SIG) dstF[(size_t)row * Nout + col] = sigmoidf_(val);
      if (EP == EP_BF16) dstB[(size_t)row * Nout + col] = (bf16_t)val;
      if (EP == EP_ADD) {
        int drow = PERM ? (((row & 255) << 8) | (row >> 8)) : row;
        dstF[(size_t)drow * Nout + col] += val;
      }
      if (EP == EP_GADD) dstF[(size_t)row * Nout + col] += val * gate[(size_t)row * Nout + col];
    }
  };
  emit(acc00, 0, 0);
  emit(acc01, 0, 16);
  emit(acc10, 16, 0);
  emit(acc11, 16, 16);
}

// ---------------------------------------------------------------- dual GEMM (proj+gate fused / transition hidden)
#define MODE_TRI 0
#define MODE_TRANS 1
template <int MODE>
__global__ __launch_bounds__(256) void k_dual(const bf16_t* __restrict__ X,
                                              const float* __restrict__ WA,
                                              const float* __restrict__ WB,
                                              const float* __restrict__ mask,
                                              bf16_t* __restrict__ dA,
                                              bf16_t* __restrict__ dB,
                                              int Nout, int K) {
  __shared__ bf16_t As[128][48];
  __shared__ bf16_t BsA[64][48];
  __shared__ bf16_t BsB[64][48];
  int r0 = blockIdx.x * 128, n0 = blockIdx.y * 64;
  int tid = threadIdx.x, wave = tid >> 5, ln = tid & 31;
  int wy = (wave >> 1) * 32, wx = (wave & 1) * 32;
  v8f aA00 = {}, aA01 = {}, aA10 = {}, aA11 = {};
  v8f aB00 = {}, aB01 = {}, aB10 = {}, aB11 = {};
  for (int k0 = 0; k0 < K; k0 += 32) {
    __syncthreads();
    {
      int m = tid >> 1, seg = (tid & 1) * 16;
      const bf16_t* g = X + (size_t)(r0 + m) * K + k0 + seg;
      async_copy_b128(&As[m][seg], g);
      async_copy_b128(&As[m][seg + 8], g + 8);
    }
    {
      int n = tid >> 2, seg = (tid & 3) * 8;
      const float* wa = WA + (size_t)(n0 + n) * K + k0 + seg;
      float4 w0 = *(const float4*)wa;
      float4 w1 = *(const float4*)(wa + 4);
      bf16_t t8[8] = {(bf16_t)w0.x, (bf16_t)w0.y, (bf16_t)w0.z, (bf16_t)w0.w,
                      (bf16_t)w1.x, (bf16_t)w1.y, (bf16_t)w1.z, (bf16_t)w1.w};
      *(uint4*)&BsA[n][seg] = *(uint4*)t8;
      const float* wb = WB + (size_t)(n0 + n) * K + k0 + seg;
      float4 v0 = *(const float4*)wb;
      float4 v1 = *(const float4*)(wb + 4);
      bf16_t s8[8] = {(bf16_t)v0.x, (bf16_t)v0.y, (bf16_t)v0.z, (bf16_t)v0.w,
                      (bf16_t)v1.x, (bf16_t)v1.y, (bf16_t)v1.z, (bf16_t)v1.w};
      *(uint4*)&BsB[n][seg] = *(uint4*)s8;
    }
    async_wait();
    __syncthreads();
    v16bf a0 = frag_a(&As[0][0], 48, wy, 0);
    v16bf a1 = frag_a(&As[0][0], 48, wy + 16, 0);
    v16bf bA0 = frag_b(&BsA[0][0], 48, wx, 0);
    v16bf bA1 = frag_b(&BsA[0][0], 48, wx + 16, 0);
    v16bf bB0 = frag_b(&BsB[0][0], 48, wx, 0);
    v16bf bB1 = frag_b(&BsB[0][0], 48, wx + 16, 0);
    aA00 = wmma_bf16(a0, bA0, aA00);
    aA01 = wmma_bf16(a0, bA1, aA01);
    aA10 = wmma_bf16(a1, bA0, aA10);
    aA11 = wmma_bf16(a1, bA1, aA11);
    aB00 = wmma_bf16(a0, bB0, aB00);
    aB01 = wmma_bf16(a0, bB1, aB01);
    aB10 = wmma_bf16(a1, bB0, aB10);
    aB11 = wmma_bf16(a1, bB1, aB11);
  }
  int nloc = ln & 15, mh = (ln >> 4) * 8;
  auto emit = [&](const v8f& pa, const v8f& pb, int mt, int nt) {
    int col = n0 + wx + nt + nloc;
#pragma unroll
    for (int r = 0; r < 8; ++r) {
      int row = r0 + wy + mt + mh + r;
      if (MODE == MODE_TRI) {
        float val = pa[r] * mask[row] * sigmoidf_(pb[r]);
        int cch = col >> 1;
        if (col & 1) dB[(size_t)cch * RN + row] = (bf16_t)val;  // b channels (odd cols)
        else         dA[(size_t)cch * RN + row] = (bf16_t)val;  // a channels (even cols)
      } else {
        float a = pa[r];
        dA[(size_t)row * Nout + col] = (bf16_t)(a * sigmoidf_(a) * pb[r]);  // silu(a)*b
      }
    }
  };
  emit(aA00, aB00, 0, 0);
  emit(aA01, aB01, 0, 16);
  emit(aA10, aB10, 16, 0);
  emit(aA11, aB11, 16, 16);
}

// ---------------------------------------------------------------- triangle einsum, channel-major batch GEMM
// OUT:  y_c[i,j] = sum_k AT[c][i][k] * BT[c][j][k]
// IN :  y_c[i,j] = sum_k BT[c][k][i] * AT[c][k][j]
template <int INCOMING>
__global__ __launch_bounds__(256) void k_einsum(const bf16_t* __restrict__ AT,
                                                const bf16_t* __restrict__ BT,
                                                float* __restrict__ Y) {
  __shared__ bf16_t As[128][48];
  __shared__ bf16_t Bs[64][48];
  int i0 = blockIdx.x * 128, j0 = blockIdx.y * 64, c = blockIdx.z;
  const bf16_t* Abase = AT + (size_t)c * RN;
  const bf16_t* Bbase = BT + (size_t)c * RN;
  int tid = threadIdx.x, wave = tid >> 5, ln = tid & 31;
  int wy = (wave >> 1) * 32, wx = (wave & 1) * 32;
  v8f acc00 = {}, acc01 = {}, acc10 = {}, acc11 = {};
  for (int k0 = 0; k0 < NSEQ; k0 += 32) {
    __syncthreads();
    if (INCOMING == 0) {
      int m = tid >> 1, seg = (tid & 1) * 16;
      const bf16_t* g = Abase + (size_t)(i0 + m) * NSEQ + k0 + seg;
      async_copy_b128(&As[m][seg], g);
      async_copy_b128(&As[m][seg + 8], g + 8);
      int n = tid >> 2, sg = (tid & 3) * 8;
      async_copy_b128(&Bs[n][sg], Bbase + (size_t)(j0 + n) * NSEQ + k0 + sg);
    } else {
      int kk = tid >> 3, g = tid & 7;
      const bf16_t* arow = Bbase + (size_t)(k0 + kk) * NSEQ + i0 + g * 16;
      bf16_t tmp[16];
      *(uint4*)tmp = *(const uint4*)arow;
      *(uint4*)(tmp + 8) = *(const uint4*)(arow + 8);
#pragma unroll
      for (int u = 0; u < 16; ++u) As[g * 16 + u][kk] = tmp[u];
      const bf16_t* brow = Abase + (size_t)(k0 + kk) * NSEQ + j0 + g * 8;
      bf16_t tb[8];
      *(uint4*)tb = *(const uint4*)brow;
#pragma unroll
      for (int u = 0; u < 8; ++u) Bs[g * 8 + u][kk] = tb[u];
    }
    async_wait();
    __syncthreads();
    v16bf a0 = frag_a(&As[0][0], 48, wy, 0);
    v16bf a1 = frag_a(&As[0][0], 48, wy + 16, 0);
    v16bf b0 = frag_b(&Bs[0][0], 48, wx, 0);
    v16bf b1 = frag_b(&Bs[0][0], 48, wx + 16, 0);
    acc00 = wmma_bf16(a0, b0, acc00);
    acc01 = wmma_bf16(a0, b1, acc01);
    acc10 = wmma_bf16(a1, b0, acc10);
    acc11 = wmma_bf16(a1, b1, acc11);
  }
  int nloc = ln & 15, mh = (ln >> 4) * 8;
  auto emit = [&](const v8f& acc, int mt, int nt) {
#pragma unroll
    for (int r = 0; r < 8; ++r)
      Y[((size_t)(i0 + wy + mt + mh + r) * NSEQ + (j0 + wx + nt + nloc)) * CDIM + c] = acc[r];
  };
  emit(acc00, 0, 0);
  emit(acc01, 0, 16);
  emit(acc10, 16, 0);
  emit(acc11, 16, 16);
}

// ---------------------------------------------------------------- fused attention per (b, h, q-tile of 32)
__global__ __launch_bounds__(256) void k_attn(const bf16_t* __restrict__ Q,
                                              const bf16_t* __restrict__ Kb,
                                              const bf16_t* __restrict__ V,
                                              const float* __restrict__ bias,
                                              const float* __restrict__ mask,
                                              float* __restrict__ O) {
  __shared__ float Sl[32][264];
  __shared__ bf16_t Vs[32][272];   // [d][k]
  __shared__ bf16_t Qs[32][40];
  __shared__ bf16_t Ks[64][40];
  int q0 = blockIdx.x * 32, h = blockIdx.y, b = blockIdx.z;
  int tid = threadIdx.x, wave = tid >> 5, ln = tid & 31;
  {
    int q = tid >> 3, du = (tid & 7) * 4;
    *(uint2*)&Qs[q][du] = *(const uint2*)(Q + ((size_t)(b * NSEQ + q0 + q)) * CDIM + h * 32 + du);
  }
  {
    int d = tid & 31, kbase = (tid >> 5) * 32;
    for (int kk = 0; kk < 32; ++kk)
      Vs[d][kbase + kk] = V[((size_t)(b * NSEQ + kbase + kk)) * CDIM + h * 32 + d];
  }
  int sq = (wave & 1) * 16, sk = (wave >> 1) * 16;
  for (int kp = 0; kp < 4; ++kp) {
    __syncthreads();
    {
      int kr = tid >> 2, du = (tid & 3) * 8;
      async_copy_b128(&Ks[kr][du],
                      Kb + ((size_t)(b * NSEQ + kp * 64 + kr)) * CDIM + h * 32 + du);
    }
    async_wait();
    __syncthreads();
    v16bf a = frag_a(&Qs[0][0], 40, sq, 0);
    v16bf bb = frag_b(&Ks[0][0], 40, sk, 0);
    v8f acc = {};
    acc = wmma_bf16(a, bb, acc);
    int nloc = ln & 15, mh = (ln >> 4) * 8;
#pragma unroll
    for (int r = 0; r < 8; ++r) Sl[sq + mh + r][kp * 64 + sk + nloc] = acc[r];
  }
  __syncthreads();
  // softmax: wave handles 4 rows
#pragma unroll
  for (int rr = 0; rr < 4; ++rr) {
    int row = wave * 4 + rr;
    int q = q0 + row;
    float vals[8];
    float mx = -1e30f;
#pragma unroll
    for (int u = 0; u < 8; ++u) {
      int col = ln + u * 32;
      float val = Sl[row][col] + bias[(size_t)h * RN + (size_t)q * NSEQ + col];
      val = (mask[(size_t)col * NSEQ + b] > 0.5f) ? val : -1e9f;
      vals[u] = val;
      mx = fmaxf(mx, val);
    }
#pragma unroll
    for (int off = 16; off > 0; off >>= 1) mx = fmaxf(mx, __shfl_xor(mx, off, 32));
    float sum = 0.f;
#pragma unroll
    for (int u = 0; u < 8; ++u) { vals[u] = __expf(vals[u] - mx); sum += vals[u]; }
#pragma unroll
    for (int off = 16; off > 0; off >>= 1) sum += __shfl_xor(sum, off, 32);
    float inv = 1.0f / (sum + 1e-20f);
#pragma unroll
    for (int u = 0; u < 8; ++u) Sl[row][ln + u * 32] = vals[u] * inv;
  }
  __syncthreads();
  if (wave < 4) {  // P * V : 32q x 32d in 4 wave tiles
    int oq = (wave & 1) * 16, od = (wave >> 1) * 16;
    v8f acc = {};
    for (int kt = 0; kt < 8; ++kt) {
      v16bf a;
      int m = oq + (ln & 15);
      int ks = kt * 32 + ((ln >> 4) * 8);
#pragma unroll
      for (int t = 0; t < 8; ++t) a[t] = (bf16_t)Sl[m][ks + t];
#pragma unroll
      for (int t = 0; t < 8; ++t) a[8 + t] = (bf16_t)Sl[m][ks + 16 + t];
      v16bf bb = frag_b(&Vs[0][0], 272, od, kt * 32);
      acc = wmma_bf16(a, bb, acc);
    }
    int nloc = ln & 15, mh = (ln >> 4) * 8;
#pragma unroll
    for (int r = 0; r < 8; ++r)
      O[((size_t)(b * NSEQ + q0 + oq + mh + r)) * CDIM + h * 32 + od + nloc] = acc[r];
  }
}

// ================================================================ host side
static constexpr size_t SZB = (size_t)RN * CDIM * 2;  // bf16 act-sized buffer
static constexpr size_t SZF = (size_t)RN * CDIM * 4;  // f32 act-sized buffer
static constexpr size_t O_XB = 0;
static constexpr size_t O_B1 = O_XB + SZB;
static constexpr size_t O_B2 = O_B1 + SZB;
static constexpr size_t O_B3 = O_B2 + SZB;
static constexpr size_t O_B4 = O_B3 + SZB;
static constexpr size_t O_F1 = O_B4 + SZB;
static constexpr size_t O_F2 = O_F1 + SZF;
static constexpr size_t O_BIAS = O_F2 + SZF;

// param order inside tri dict (sorted): Wg, Wgl, Wo, Wp, ln1_b, ln1_s, ln2_b, ln2_s
static void run_tri(float* act, const float* mask, void* const* p, int incoming,
                    char* ws, hipStream_t stream) {
  const float *Wg = (const float*)p[0], *Wgl = (const float*)p[1];
  const float *Wo = (const float*)p[2], *Wp = (const float*)p[3];
  const float *ln1_b = (const float*)p[4], *ln1_s = (const float*)p[5];
  const float *ln2_b = (const float*)p[6], *ln2_s = (const float*)p[7];
  bf16_t* XB = (bf16_t*)(ws + O_XB);
  bf16_t* AT = (bf16_t*)(ws + O_B1);
  bf16_t* BT = (bf16_t*)(ws + O_B2);
  bf16_t* YB = (bf16_t*)(ws + O_B3);
  float* Y = (float*)(ws + O_F1);
  float* GL = (float*)(ws + O_F2);
  k_ln<<<RN / 8, 256, 0, stream>>>(act, ln1_s, ln1_b, XB);
  k_dual<MODE_TRI><<<dim3(RN / 128, 4), 256, 0, stream>>>(XB, Wp, Wg, mask, AT, BT, 256, CDIM);
  k_gemm<0, EP_SIG, 0><<<dim3(RN / 128, 2), 256, 0, stream>>>(XB, Wgl, GL, nullptr, nullptr, CDIM, CDIM, 1.0f);
  if (incoming)
    k_einsum<1><<<dim3(2, 4, CDIM), 256, 0, stream>>>(AT, BT, Y);
  else
    k_einsum<0><<<dim3(2, 4, CDIM), 256, 0, stream>>>(AT, BT, Y);
  k_ln<<<RN / 8, 256, 0, stream>>>(Y, ln2_s, ln2_b, YB);
  k_gemm<0, EP_GADD, 0><<<dim3(RN / 128, 2), 256, 0, stream>>>(YB, Wo, act, nullptr, GL, CDIM, CDIM, 1.0f);
}

// param order inside att dict (sorted): Wb, Wgate, Wk, Wout, Wq, Wv, ln_b, ln_s
static void run_att(float* act, const float* mask, void* const* p, int T,
                    char* ws, hipStream_t stream) {
  const float *Wb = (const float*)p[0], *Wgate = (const float*)p[1];
  const float *Wk = (const float*)p[2], *Wout = (const float*)p[3];
  const float *Wq = (const float*)p[4], *Wv = (const float*)p[5];
  const float *ln_b = (const float*)p[6], *ln_s = (const float*)p[7];
  bf16_t* XB = (bf16_t*)(ws + O_XB);
  bf16_t* Qb = (bf16_t*)(ws + O_B1);
  bf16_t* Kbuf = (bf16_t*)(ws + O_B2);
  bf16_t* XT = (bf16_t*)(ws + O_B3);
  bf16_t* Vb = (bf16_t*)(ws + O_B4);
  float* Obuf = (float*)(ws + O_F1);
  float* Gate = (float*)(ws + O_F2);
  float* Bias = (float*)(ws + O_BIAS);
  k_ln<<<RN / 8, 256, 0, stream>>>(act, ln_s, ln_b, XB);
  k_bias<<<(RN * 4) / 256, 256, 0, stream>>>(XB, Wb, Bias);
  const bf16_t* src = XB;
  if (T) {
    k_transpose128<<<RN / 8, 256, 0, stream>>>(XB, XT);
    src = XT;
  }
  const float qscale = 0.17677669529663687f;  // 1/sqrt(32)
  k_gemm<1, EP_BF16, 0><<<dim3(RN / 128, 2), 256, 0, stream>>>(src, Wq, nullptr, Qb, nullptr, CDIM, CDIM, qscale);
  k_gemm<1, EP_BF16, 0><<<dim3(RN / 128, 2), 256, 0, stream>>>(src, Wk, nullptr, Kbuf, nullptr, CDIM, CDIM, 1.0f);
  k_gemm<1, EP_BF16, 0><<<dim3(RN / 128, 2), 256, 0, stream>>>(src, Wv, nullptr, Vb, nullptr, CDIM, CDIM, 1.0f);
  k_gemm<1, EP_SIG, 0><<<dim3(RN / 128, 2), 256, 0, stream>>>(src, Wgate, Gate, nullptr, nullptr, CDIM, CDIM, 1.0f);
  k_attn<<<dim3(8, 4, NSEQ), 256, 0, stream>>>(Qb, Kbuf, Vb, Bias, mask, Obuf);
  k_og<<<(RN * CDIM / 4) / 256, 256, 0, stream>>>(Obuf, Gate, Qb);  // reuse Qb as O*gate
  if (T)
    k_gemm<0, EP_ADD, 1><<<dim3(RN / 128, 2), 256, 0, stream>>>(Qb, Wout, act, nullptr, nullptr, CDIM, CDIM, 1.0f);
  else
    k_gemm<0, EP_ADD, 0><<<dim3(RN / 128, 2), 256, 0, stream>>>(Qb, Wout, act, nullptr, nullptr, CDIM, CDIM, 1.0f);
}

// param order inside trans dict (sorted): W1, W2, ln_b, ln_s
static void run_trans(float* act, void* const* p, char* ws, hipStream_t stream) {
  const float* W1 = (const float*)p[0];
  const float* W2 = (const float*)p[1];
  const float* ln_b = (const float*)p[2];
  const float* ln_s = (const float*)p[3];
  bf16_t* XB = (bf16_t*)(ws + O_XB);
  bf16_t* Hb = (bf16_t*)(ws + O_F1);  // R x 512 bf16 spans F1+F2
  k_ln<<<RN / 8, 256, 0, stream>>>(act, ln_s, ln_b, XB);
  k_dual<MODE_TRANS><<<dim3(RN / 128, 8), 256, 0, stream>>>(XB, W1, W1 + 512 * CDIM, nullptr, Hb, nullptr, 512, CDIM);
  k_gemm<0, EP_ADD, 0><<<dim3(RN / 128, 2), 256, 0, stream>>>(Hb, W2, act, nullptr, nullptr, CDIM, 512, 1.0f);
}

extern "C" void kernel_launch(void* const* d_in, const int* in_sizes, int n_in,
                              void* d_out, int out_size, void* d_ws, size_t ws_size,
                              hipStream_t stream) {
  (void)in_sizes; (void)n_in; (void)out_size; (void)ws_size;
  // d_in order (jax pytree, dict keys sorted):
  //  0: act, 1: pair_mask,
  //  2..9  : att1  {Wb,Wgate,Wk,Wout,Wq,Wv,ln_b,ln_s}
  // 10..17 : att2  {same}
  // 18..21 : trans {W1,W2,ln_b,ln_s}
  // 22..29 : tri_in  {Wg,Wgl,Wo,Wp,ln1_b,ln1_s,ln2_b,ln2_s}
  // 30..37 : tri_out {same}
  const float* act_in = (const float*)d_in[0];
  const float* mask = (const float*)d_in[1];
  float* act = (float*)d_out;
  char* ws = (char*)d_ws;

  hipMemcpyAsync(act, act_in, (size_t)RN * CDIM * sizeof(float), hipMemcpyDeviceToDevice, stream);

  run_tri(act, mask, d_in + 30, /*incoming=*/0, ws, stream);  // tri_out: 'ikc,jkc->ijc'
  run_tri(act, mask, d_in + 22, /*incoming=*/1, ws, stream);  // tri_in : 'kjc,kic->ijc'
  run_att(act, mask, d_in + 2, /*transpose=*/0, ws, stream);  // att1
  run_att(act, mask, d_in + 10, /*transpose=*/1, ws, stream); // att2
  run_trans(act, d_in + 18, ws, stream);                      // transition
}
